// MyModule_52699248722431
// MI455X (gfx1250) — compile-verified, matrix operand order checked by reading, not run on hardware
//
#include <hip/hip_runtime.h>
#include <math.h>

// ---------- constants ----------
#define BB 4
#define SS 2048
#define DD 1024
#define HH 16
#define HD 64
// M = BB*SS = 8192

typedef __attribute__((ext_vector_type(16))) __bf16 v16bf;
typedef __attribute__((ext_vector_type(8)))  float  v8f;

#define TPK 72   // LDS K-pitch (bf16 elems) for 64-wide tiles: 144B rows, 16B aligned
#define PPK 136  // LDS pitch for 128-wide P tiles: 272B rows, 16B aligned

// ---------- small helpers ----------
__device__ __forceinline__ unsigned short f2bf(float f) {
  unsigned u = __float_as_uint(f);
  unsigned r = u + 0x7fffu + ((u >> 16) & 1u);   // round-to-nearest-even
  return (unsigned short)(r >> 16);
}

__device__ __forceinline__ unsigned lds_off(const void* p) {
  // generic pointer into LDS: low 32 bits are the LDS offset
  return (unsigned)(unsigned long long)(size_t)p;
}

__device__ __forceinline__ void async_copy_b128(unsigned loff, const void* g) {
  unsigned long long ga = (unsigned long long)(size_t)g;
  asm volatile("global_load_async_to_lds_b128 %0, %1, off"
               :: "v"(loff), "v"(ga) : "memory");
}

// Wait until at most N async copies remain outstanding (in-order retirement
// means the oldest ones are the ones that completed).
template <int N>
__device__ __forceinline__ void wait_async_n() {
  asm volatile("s_wait_asynccnt %0" :: "n"(N) : "memory");
}

__device__ __forceinline__ v8f wmma_bf16(v16bf a, v16bf b, v8f c) {
  return __builtin_amdgcn_wmma_f32_16x16x32_bf16(false, a, false, b,
                                                 (short)0, c, false, false);
}

// Load a 16x32 A (or 16-col B) fragment from an LDS tile stored row-major
// [row][K] with pitch `pitch` (bf16 elems). Per the CDNA5 16-bit layout:
//  lane 0..15  -> K in {k0+0..7, k0+16..23}
//  lane 16..31 -> K in {k0+8..15, k0+24..31}
__device__ __forceinline__ v16bf frag_rowmajor(const unsigned short* lds,
                                               int row, int k0, int pitch,
                                               int laneHi) {
  union { unsigned int u[8]; v16bf v; } r;
  const unsigned short* p = lds + row * pitch + k0 + (laneHi ? 8 : 0);
  #pragma unroll
  for (int i = 0; i < 4; i++)
    r.u[i] = *(const unsigned int*)(p + 2 * i);
  #pragma unroll
  for (int i = 0; i < 4; i++)
    r.u[4 + i] = *(const unsigned int*)(p + 16 + 2 * i);
  return r.v;
}

// B fragment where the LDS tile is stored [K][N] (natural V layout):
// lane's column n fixed, K follows the standard pattern.
__device__ __forceinline__ v16bf frag_colmajor(const unsigned short* lds,
                                               int col, int k0, int pitch,
                                               int laneHi) {
  union { unsigned short h[16]; v16bf v; } r;
  int base = k0 + (laneHi ? 8 : 0);
  #pragma unroll
  for (int e = 0; e < 16; e++) {
    int k = base + (e & 7) + ((e >> 3) << 4);
    r.h[e] = lds[k * pitch + col];
  }
  return r.v;
}

__device__ __forceinline__ float redmax16(float v) {
  #pragma unroll
  for (int m = 1; m < 16; m <<= 1) v = fmaxf(v, __shfl_xor(v, m, 32));
  return v;
}
__device__ __forceinline__ float redsum16(float v) {
  #pragma unroll
  for (int m = 1; m < 16; m <<= 1) v += __shfl_xor(v, m, 32);
  return v;
}

// ---------- kernel 1: f32 -> bf16 convert ----------
__global__ void cvt_bf16_kernel(const float* __restrict__ src,
                                unsigned short* __restrict__ dst, int n) {
  int i = blockIdx.x * 256 + threadIdx.x;
  if (i < n) dst[i] = f2bf(src[i]);
}

// ---------- kernel 2: depthwise conv along S ----------
// out[b,s,d] = w[d,0]*x[b,s-1,d] + w[d,1]*x[b,s,d] + w[d,2]*x[b,s+1,d]
// (only the middle column kw=1 of the 3x3 kernel contributes since W==1)
__global__ void conv_dw_kernel(const float* __restrict__ x,
                               const float* __restrict__ w,
                               float* __restrict__ xsF,
                               unsigned short* __restrict__ xsH) {
  int idx = blockIdx.x * 256 + threadIdx.x;     // over B*S*D
  int d = idx & (DD - 1);
  int s = (idx >> 10) & (SS - 1);
  float w0 = w[d * 9 + 1], w1 = w[d * 9 + 4], w2 = w[d * 9 + 7];
  float xm = (s > 0)      ? x[idx - DD] : 0.f;
  float xc = x[idx];
  float xp = (s < SS - 1) ? x[idx + DD] : 0.f;
  float r = w0 * xm + w1 * xc + w2 * xp;
  xsF[idx] = r;
  xsH[idx] = f2bf(r);
}

// ---------- kernel 3: bf16 WMMA GEMM  C = A * W^T (+epilogue) ----------
// A: [8192 x K] bf16 row-major, W: [N x K] bf16 row-major (so B[k][n]=W[n][k])
// Double-buffered async staging: copies for tile t+1 fly under tile t's WMMAs.
// mode 0: out_bf16 = C + resid_f32  (pe residual)
// mode 1: split n into 3 chunks of 1024 -> q/k/v buffers (stride 8388608 elems)
// mode 2: out_f32 = C (final projection)
__device__ __forceinline__ void gemm_stage(unsigned short* At, unsigned short* Bt,
                                           const unsigned short* A,
                                           const unsigned short* W,
                                           int m0, int n0, int K, int k0,
                                           int wave, int lane) {
  #pragma unroll
  for (int j = 0; j < 4; j++) {          // 128x64 bf16 tiles, 8 copies/wave
    int row = wave * 16 + j * 4 + (lane >> 3);
    int c = (lane & 7) * 8;              // 8 bf16 = 16B per lane
    async_copy_b128(lds_off(&At[row * TPK + c]),
                    A + (size_t)(m0 + row) * K + k0 + c);
    async_copy_b128(lds_off(&Bt[row * TPK + c]),
                    W + (size_t)(n0 + row) * K + k0 + c);
  }
}

__global__ void __launch_bounds__(256)
gemm_bf16_kernel(const unsigned short* __restrict__ A,
                 const unsigned short* __restrict__ W,
                 int N, int K, int mode,
                 const float* __restrict__ resid,
                 unsigned short* __restrict__ outH,
                 float* __restrict__ outF) {
  __shared__ unsigned short At[2][128 * TPK];
  __shared__ unsigned short Bt[2][128 * TPK];
  int tid = threadIdx.x, wave = tid >> 5, lane = tid & 31;
  int l15 = lane & 15, laneHi = lane >> 4;
  int m0 = blockIdx.x * 128, n0 = blockIdx.y * 128;
  int waveM = (wave >> 2) * 64, waveN = (wave & 3) * 32;

  v8f acc[4][2];
  #pragma unroll
  for (int i = 0; i < 4; i++)
    #pragma unroll
    for (int j = 0; j < 2; j++)
      #pragma unroll
      for (int e = 0; e < 8; e++) acc[i][j][e] = 0.f;

  const int nk = K >> 6;                 // K-steps of 64
  gemm_stage(At[0], Bt[0], A, W, m0, n0, K, 0, wave, lane);

  for (int it = 0; it < nk; it++) {
    if (it + 1 < nk) {                   // prefetch next tile under compute
      gemm_stage(At[(it + 1) & 1], Bt[(it + 1) & 1], A, W, m0, n0, K,
                 (it + 1) << 6, wave, lane);
      wait_async_n<8>();                 // current tile resident
    } else {
      wait_async_n<0>();
    }
    __syncthreads();

    const unsigned short* Ac = At[it & 1];
    const unsigned short* Bc = Bt[it & 1];
    #pragma unroll
    for (int kc = 0; kc < 64; kc += 32) {
      v16bf bf0 = frag_rowmajor(Bc, waveN + l15,      kc, TPK, laneHi);
      v16bf bf1 = frag_rowmajor(Bc, waveN + 16 + l15, kc, TPK, laneHi);
      #pragma unroll
      for (int mi = 0; mi < 4; mi++) {
        v16bf af = frag_rowmajor(Ac, waveM + mi * 16 + l15, kc, TPK, laneHi);
        acc[mi][0] = wmma_bf16(af, bf0, acc[mi][0]);
        acc[mi][1] = wmma_bf16(af, bf1, acc[mi][1]);
      }
    }
    __syncthreads();                     // all reads done before next overwrite
  }

  // epilogue: C-layout -> lane holds row r+8*laneHi, col l15 of each 16x16
  #pragma unroll
  for (int mi = 0; mi < 4; mi++)
    #pragma unroll
    for (int nj = 0; nj < 2; nj++)
      #pragma unroll
      for (int r = 0; r < 8; r++) {
        int m = m0 + waveM + mi * 16 + r + 8 * laneHi;
        int n = n0 + waveN + nj * 16 + l15;
        float c = acc[mi][nj][r];
        if (mode == 0) {
          c += resid[(size_t)m * DD + n];
          outH[(size_t)m * DD + n] = f2bf(c);
        } else if (mode == 1) {
          int sel = n >> 10, nn = n & 1023;
          outH[(size_t)sel * 8388608 + (size_t)m * DD + nn] = f2bf(c);
        } else {
          outF[(size_t)m * DD + n] = c;
        }
      }
}

// ---------- kernel 4: flash attention per (b,h), bf16 WMMA ----------
// head h of batch b = contiguous [2048 x 64] bf16 block (raw-reshape view).
// K/V tiles double-buffered: next block's async copies fly under this block's
// WMMAs + softmax.
__device__ __forceinline__ void attn_stage_kv(unsigned short* Kt,
                                              unsigned short* Vt,
                                              const unsigned short* Kg,
                                              const unsigned short* Vg,
                                              int t, int wave, int lane) {
  #pragma unroll
  for (int j = 0; j < 4; j++) {          // 8 copies/wave per stage
    int row = wave * 16 + j * 4 + (lane >> 3);
    int c = (lane & 7) * 8;
    async_copy_b128(lds_off(&Kt[row * TPK + c]),
                    Kg + (size_t)(t * 128 + row) * HD + c);
    async_copy_b128(lds_off(&Vt[row * TPK + c]),
                    Vg + (size_t)(t * 128 + row) * HD + c);
  }
}

__global__ void __launch_bounds__(256)
attn_kernel(const unsigned short* __restrict__ qb,
            const unsigned short* __restrict__ kb,
            const unsigned short* __restrict__ vb,
            unsigned short* __restrict__ ob) {
  __shared__ unsigned short Qt[128 * TPK];
  __shared__ unsigned short Kt[2][128 * TPK];
  __shared__ unsigned short Vt[2][128 * TPK];
  __shared__ unsigned short Pl[8 * 16 * PPK];

  int tid = threadIdx.x, wave = tid >> 5, lane = tid & 31;
  int l15 = lane & 15, laneHi = lane >> 4;
  int bid = blockIdx.x;
  int qt = bid & 15, h = (bid >> 4) & 15, b = bid >> 8;
  size_t headOff = (size_t)b * (SS * DD) + (size_t)h * (SS * HD);
  const unsigned short* Qg = qb + headOff + (size_t)qt * 128 * HD;
  const unsigned short* Kg = kb + headOff;
  const unsigned short* Vg = vb + headOff;

  // stage Q tile (4 copies), then K/V block 0 (8 copies) behind it
  #pragma unroll
  for (int j = 0; j < 4; j++) {
    int row = wave * 16 + j * 4 + (lane >> 3);
    int c = (lane & 7) * 8;
    async_copy_b128(lds_off(&Qt[row * TPK + c]), Qg + (size_t)row * HD + c);
  }
  attn_stage_kv(Kt[0], Vt[0], Kg, Vg, 0, wave, lane);
  wait_async_n<8>();                     // retires the 4 Q copies (in-order)
  __syncthreads();

  v16bf aq0 = frag_rowmajor(Qt, wave * 16 + l15, 0,  TPK, laneHi);
  v16bf aq1 = frag_rowmajor(Qt, wave * 16 + l15, 32, TPK, laneHi);

  v8f O[4];
  #pragma unroll
  for (int t = 0; t < 4; t++)
    #pragma unroll
    for (int e = 0; e < 8; e++) O[t][e] = 0.f;
  float m_i[8], l_i[8];
  #pragma unroll
  for (int r = 0; r < 8; r++) { m_i[r] = -1e30f; l_i[r] = 0.f; }
  unsigned short* Pw = &Pl[wave * 16 * PPK];
  const float scale = 0.125f;            // 1/sqrt(64)

  for (int t = 0; t < 16; t++) {         // 16 key blocks of 128
    if (t + 1 < 16) {                    // prefetch next K/V under compute
      attn_stage_kv(Kt[(t + 1) & 1], Vt[(t + 1) & 1], Kg, Vg, t + 1, wave, lane);
      wait_async_n<8>();                 // current K/V resident
    } else {
      wait_async_n<0>();
    }
    __syncthreads();
    const unsigned short* Kc = Kt[t & 1];
    const unsigned short* Vc = Vt[t & 1];

    // S = Q K^T : 8 C-tiles of 16x16, K-dim = 64 (2 WMMAs each)
    v8f sc[8];
    #pragma unroll
    for (int nj = 0; nj < 8; nj++) {
      #pragma unroll
      for (int e = 0; e < 8; e++) sc[nj][e] = 0.f;
      v16bf bk0 = frag_rowmajor(Kc, nj * 16 + l15, 0,  TPK, laneHi);
      sc[nj] = wmma_bf16(aq0, bk0, sc[nj]);
      v16bf bk1 = frag_rowmajor(Kc, nj * 16 + l15, 32, TPK, laneHi);
      sc[nj] = wmma_bf16(aq1, bk1, sc[nj]);
    }

    // online softmax stats (row r+8*laneHi lives at reg r of this lane-half)
    float al[8];
    #pragma unroll
    for (int r = 0; r < 8; r++) {
      float v = sc[0][r];
      #pragma unroll
      for (int nj = 1; nj < 8; nj++) v = fmaxf(v, sc[nj][r]);
      v = redmax16(v) * scale;
      float mn = fmaxf(m_i[r], v);
      al[r] = __expf(m_i[r] - mn);
      l_i[r] *= al[r];
      m_i[r] = mn;
    }
    #pragma unroll
    for (int o = 0; o < 4; o++)
      #pragma unroll
      for (int r = 0; r < 8; r++) O[o][r] *= al[r];

    float rs[8];
    #pragma unroll
    for (int r = 0; r < 8; r++) rs[r] = 0.f;
    #pragma unroll
    for (int nj = 0; nj < 8; nj++)
      #pragma unroll
      for (int r = 0; r < 8; r++) {
        float p = __expf(sc[nj][r] * scale - m_i[r]);
        rs[r] += p;
        Pw[(r + 8 * laneHi) * PPK + nj * 16 + l15] = f2bf(p);
      }
    #pragma unroll
    for (int r = 0; r < 8; r++) l_i[r] += redsum16(rs[r]);

    // O += P * V   (P in LDS as A-layout source; V tile is [K][N] natural)
    #pragma unroll
    for (int kc = 0; kc < 128; kc += 32) {
      v16bf ap = frag_rowmajor(Pw, l15, kc, PPK, laneHi);
      #pragma unroll
      for (int ni = 0; ni < 4; ni++) {
        v16bf bv = frag_colmajor(Vc, ni * 16 + l15, kc, TPK, laneHi);
        O[ni] = wmma_bf16(ap, bv, O[ni]);
      }
    }
    __syncthreads();                     // all reads done before next overwrite
  }

  // write o[b, s', h*64+feat] (the transpose-back of the reshaped view)
  #pragma unroll
  for (int ni = 0; ni < 4; ni++)
    #pragma unroll
    for (int r = 0; r < 8; r++) {
      int row = qt * 128 + wave * 16 + r + 8 * laneHi;
      int col = h * HD + ni * 16 + l15;
      ob[((size_t)b * SS + row) * DD + col] = f2bf(O[ni][r] / l_i[r]);
    }
}

// ---------- host-side launch ----------
extern "C" void kernel_launch(void* const* d_in, const int* in_sizes, int n_in,
                              void* d_out, int out_size, void* d_ws, size_t ws_size,
                              hipStream_t stream) {
  const float* x     = (const float*)d_in[0];
  const float* convw = (const float*)d_in[1];
  const float* pew   = (const float*)d_in[2];
  const float* qkvw  = (const float*)d_in[3];
  const float* outw  = (const float*)d_in[4];

  char* ws = (char*)d_ws;
  float*          xsF   = (float*)(ws);                          // 32 MB
  unsigned short* xsH   = (unsigned short*)(ws + 33554432);      // 16 MB
  unsigned short* xs2H  = (unsigned short*)(ws + 50331648);      // 16 MB
  unsigned short* qH    = (unsigned short*)(ws + 67108864);      // 48 MB (q,k,v)
  unsigned short* kH    = qH + 8388608;
  unsigned short* vH    = qH + 16777216;
  unsigned short* oH    = (unsigned short*)(ws + 117440512);     // 16 MB
  unsigned short* peWH  = (unsigned short*)(ws + 134217728);     // 2 MB
  unsigned short* qkvWH = (unsigned short*)(ws + 136314880);     // 6 MB
  unsigned short* outWH = (unsigned short*)(ws + 142606336);     // 2 MB

  cvt_bf16_kernel<<<(1048576 + 255) / 256, 256, 0, stream>>>(pew,  peWH, 1048576);
  cvt_bf16_kernel<<<(3145728 + 255) / 256, 256, 0, stream>>>(qkvw, qkvWH, 3145728);
  cvt_bf16_kernel<<<(1048576 + 255) / 256, 256, 0, stream>>>(outw, outWH, 1048576);

  conv_dw_kernel<<<(BB * SS * DD) / 256, 256, 0, stream>>>(x, convw, xsF, xsH);

  dim3 gPE(64, 8);
  gemm_bf16_kernel<<<gPE, 256, 0, stream>>>(xsH, peWH, 1024, 1024, 0, xsF, xs2H, nullptr);

  dim3 gQKV(64, 24);
  gemm_bf16_kernel<<<gQKV, 256, 0, stream>>>(xs2H, qkvWH, 3072, 1024, 1, nullptr, qH, nullptr);

  attn_kernel<<<1024, 256, 0, stream>>>(qH, kH, vH, oH);

  dim3 gOUT(64, 8);
  gemm_bf16_kernel<<<gOUT, 256, 0, stream>>>(oH, outWH, 1024, 1024, 2, nullptr, nullptr,
                                             (float*)d_out);
}
